// MultiHeadAttentionAwareTemporalContex_qc_k1d_41601053229802
// MI455X (gfx1250) — matrix operand order repaired and checked by reading, not compiled, not run on hardware
//
#include <hip/hip_runtime.h>

#define TT 48
#define DD 128

typedef __attribute__((ext_vector_type(16))) __bf16 v16bf;
typedef __attribute__((ext_vector_type(8)))  float  v8f;

union BF16x16 { v16bf v; unsigned short u[16]; };

__device__ __forceinline__ unsigned short f2bf(float x) {
  union { __bf16 b; unsigned short u; } c;
  c.b = (__bf16)x;                       // native RNE f32->bf16 convert
  return c.u;
}

__device__ __forceinline__ int seg_lo(int t) { return t < 12 ? 0  : (t < 24 ? 12 : 24); }
__device__ __forceinline__ int seg_hi(int t) { return t < 12 ? 12 : (t < 24 ? 24 : 48); }

// ---------------- fragment loaders (layouts used by this file) ----------------
// A (16x32 bf16): lanes 0-15 -> M=lane, K 0..15;        lanes 16-31 -> M=lane-16, K 16..31
// B (32x16 bf16): lanes 0-15 -> N=lane, K 0..15;        lanes 16-31 -> N=lane-16, K 16..31
// C (16x16 f32) : lanes 0-15 -> N=lane, M=i;            lanes 16-31 -> N=lane-16, M=8+i

// plain A fragment, rows row0.., K cols k0.., row-major bf16 LDS
__device__ __forceinline__ v16bf load_a_row(const unsigned short* s, int ld, int row0, int k0) {
  const int lane = threadIdx.x & 31;
  const int m  = row0 + (lane & 15);
  const int kh = (lane & 16) ? 16 : 0;
  const unsigned short* p = s + m * ld + k0;
  BF16x16 a;
#pragma unroll
  for (int i = 0; i < 8; ++i) a.u[i]     = p[kh + i];
#pragma unroll
  for (int i = 0; i < 8; ++i) a.u[8 + i] = p[kh + 8 + i];
  return a.v;
}

// A fragment with temporal shift. Invalid rows (shifted source leaves the row's
// segment) are redirected to the all-zero padding row 48 -> loads stay
// unconditional ds_load_b128; only the ADDRESS is selected (no exec juggling).
__device__ __forceinline__ v16bf load_a_shift(const unsigned short* s, int row0, int k0, int shift) {
  const int lane = threadIdx.x & 31;
  const int m  = row0 + (lane & 15);
  const int kh = (lane & 16) ? 16 : 0;
  int src = m + shift;
  const bool ok = (src >= seg_lo(m)) && (src < seg_hi(m));
  src = ok ? src : TT;                       // row 48 = zeros
  const unsigned short* p = s + src * DD + k0;
  BF16x16 a;
#pragma unroll
  for (int i = 0; i < 8; ++i) a.u[i]     = p[kh + i];
#pragma unroll
  for (int i = 0; i < 8; ++i) a.u[8 + i] = p[kh + 8 + i];
  return a.v;
}

// A fragment with only K=0..15 valid (DK=16 padded to 32): for Q·K^T
__device__ __forceinline__ v16bf load_a_k16(const unsigned short* s, int ld, int row0, int col0) {
  const int lane = threadIdx.x & 31;
  const int m  = row0 + (lane & 15);
  const int kh = (lane & 16) ? 8 : 0;
  const unsigned short* p = s + m * ld + col0;
  BF16x16 a;
#pragma unroll
  for (int i = 0; i < 8; ++i) a.u[i]     = kh ? 0 : p[i];
#pragma unroll
  for (int i = 0; i < 8; ++i) a.u[8 + i] = kh ? 0 : p[8 + i];
  return a.v;
}

// B fragment from pre-swizzled (fragment-major) weights in global/L2:
// element (k0+kh+i, n0+n) lives at (((k0/32)*8 + n0/16)*512 + n*32 + (kh?16:0) + i)
__device__ __forceinline__ v16bf load_b_g(const unsigned short* __restrict__ w, int kc, int nt) {
  const int lane = threadIdx.x & 31;
  const unsigned short* p = w + (((kc << 3) + nt) << 9) + ((lane & 15) << 5) + ((lane & 16) ? 16 : 0);
  BF16x16 b;
#pragma unroll
  for (int i = 0; i < 16; ++i) b.u[i] = p[i];   // 32B contiguous -> 2x global_load_b128
  return b.v;
}

// B = K_h^T from per-head [s][32] LDS (dk 16..31 zero-padded): unconditional
__device__ __forceinline__ v16bf load_b_kT(const unsigned short* skt_h, int n0) {
  const int lane = threadIdx.x & 31;
  const unsigned short* p = skt_h + ((n0 + (lane & 15)) << 5) + ((lane & 16) ? 16 : 0);
  BF16x16 b;
#pragma unroll
  for (int i = 0; i < 16; ++i) b.u[i] = p[i];    // 32B contiguous
  return b.v;
}

// B = V_h from per-head transposed [n][k(64 padded)] LDS
__device__ __forceinline__ v16bf load_b_vT(const unsigned short* svt_h, int k0) {
  const int lane = threadIdx.x & 31;
  const unsigned short* p = svt_h + ((lane & 15) << 6) + k0 + ((lane & 16) ? 16 : 0);
  BF16x16 b;
#pragma unroll
  for (int i = 0; i < 16; ++i) b.u[i] = p[i];    // 32B contiguous
  return b.v;
}

// A fragment from padded P (48x64 row-major, ld=64): unconditional contiguous
__device__ __forceinline__ v16bf load_a_p64(const unsigned short* s, int row0, int k0) {
  const int lane = threadIdx.x & 31;
  const int m  = row0 + (lane & 15);
  const int kh = (lane & 16) ? 16 : 0;
  const unsigned short* p = s + (m << 6) + k0;
  BF16x16 a;
#pragma unroll
  for (int i = 0; i < 8; ++i) a.u[i]     = p[kh + i];
#pragma unroll
  for (int i = 0; i < 8; ++i) a.u[8 + i] = p[kh + 8 + i];
  return a.v;
}

// store C fragment as bf16 into row-major LDS (column scatter, unavoidable)
__device__ __forceinline__ void store_c_bf(unsigned short* s, int ld, int row0, int n0, v8f c) {
  const int lane = threadIdx.x & 31;
  const int n  = n0 + (lane & 15);
  const int mh = (lane & 16) ? 8 : 0;
#pragma unroll
  for (int i = 0; i < 8; ++i) s[(row0 + mh + i) * ld + n] = f2bf(c[i]);
}

// store C fragment transposed into per-head V^T [n][64]: 16B contiguous per lane
__device__ __forceinline__ void store_c_vT(unsigned short* svt_h, int row0, v8f c) {
  const int lane = threadIdx.x & 31;
  const int n  = lane & 15;
  const int mh = (lane & 16) ? 8 : 0;
  union { unsigned short u[8]; uint4 q; } t;
#pragma unroll
  for (int i = 0; i < 8; ++i) t.u[i] = f2bf(c[i]);
  *(uint4*)(svt_h + (n << 6) + row0 + mh) = t.q;   // ds_store_b128
}

// c[3] = bias + sum_j shift_{SHIFT0+j}(sX) @ Wsw[j]  (B fragments from L2)
template<int NSHIFT, int SHIFT0>
__device__ __forceinline__ void proj(const unsigned short* __restrict__ Wsw,
                                     const float* __restrict__ bias,
                                     const unsigned short* sX, int wave, int lane, v8f c[3]) {
  const float bv = bias[wave * 16 + (lane & 15)];
#pragma unroll
  for (int mt = 0; mt < 3; ++mt)
#pragma unroll
    for (int i = 0; i < 8; ++i) c[mt][i] = bv;
#pragma unroll
  for (int j = 0; j < NSHIFT; ++j) {
#pragma unroll
    for (int kc = 0; kc < 4; ++kc) {
      v16bf bf = load_b_g(Wsw + (j << 14), kc, wave);
#pragma unroll
      for (int mt = 0; mt < 3; ++mt) {
        v16bf af = load_a_shift(sX, mt * 16, kc * 32, SHIFT0 + j);
        c[mt] = __builtin_amdgcn_wmma_f32_16x16x32_bf16(false, af, false, bf, (short)0, c[mt], false, false);
      }
    }
  }
}

// ---------------- pre-kernel: swizzle all weights to fragment-major bf16 ----------------
__global__ __launch_bounds__(256)
void swizzle_weights_kernel(const float* __restrict__ qw, const float* __restrict__ kw,
                            const float* __restrict__ vw, const float* __restrict__ ow,
                            unsigned short* __restrict__ wsw) {
  const int e = blockIdx.x * 256 + threadIdx.x;           // 8 matrices * 16384
  if (e >= 8 * DD * DD) return;
  const int m = e >> 14, r = e & 16383;
  const float* src = (m < 3) ? (qw + m * DD * DD)
                   : (m < 6) ? (kw + (m - 3) * DD * DD)
                   : (m == 6) ? vw : ow;
  const int k = r >> 7, n = r & 127;
  const int idx = (((k >> 5) * 8 + (n >> 4)) << 9) + ((n & 15) << 5) + (((k >> 4) & 1) << 4) + (k & 15);
  wsw[(m << 14) + idx] = f2bf(src[r]);
}

// ---------------- fused attention kernel: one workgroup per (b, n) ----------------
__global__ __launch_bounds__(256)
void fused_attn_kernel(const float* __restrict__ q_in, const float* __restrict__ k_in,
                       const float* __restrict__ v_in,
                       const unsigned short* __restrict__ wsw,
                       const float* __restrict__ qb, const float* __restrict__ kb,
                       const float* __restrict__ vb, const float* __restrict__ ob,
                       float* __restrict__ out) {
  // 100KB pool: sX (conv staging + zero row) overlaid by sP (attention phase)
  __shared__ __align__(16) unsigned short pool[51200];
  unsigned short* sX  = pool;              // 49*128 (row 48 = zeros)   conv phase
  unsigned short* sP  = pool;              // 8 * 48*64 = 24576         attention phase
  unsigned short* sQ  = pool + 24576;      // 48*128 row-major (q, later x)
  unsigned short* sKT = pool + 30720;      // 8 * 48*32 per-head [s][dk], dk padded to 32
  unsigned short* sVT = pool + 43008;      // 8 * 16*64 per-head [n][k], k padded to 64

  const int lane = threadIdx.x & 31;
  const int wave = threadIdx.x >> 5;       // 0..7 : output N-tile == head id
  const long base = (long)blockIdx.x * TT * DD;

  const unsigned short* wsQ = wsw;
  const unsigned short* wsK = wsw + 3 * DD * DD;
  const unsigned short* wsV = wsw + 6 * DD * DD;
  const unsigned short* wsO = wsw + 7 * DD * DD;

  // one-time zero fills (regions never overwritten during their live phase)
  for (int e = threadIdx.x; e < DD / 8; e += 256)                // sX zero row 48
    *(uint4*)(sX + TT * DD + e * 8) = uint4{0, 0, 0, 0};
  for (int e = threadIdx.x; e < 8 * TT * 2; e += 256)            // sKT dk=16..31 pad
    *(uint4*)(sKT + (e >> 1) * 32 + 16 + (e & 1) * 8) = uint4{0, 0, 0, 0};
  for (int e = threadIdx.x; e < 8 * 16 * 2; e += 256)            // sVT k=48..63 pad
    *(uint4*)(sVT + (e >> 1) * 64 + 48 + (e & 1) * 8) = uint4{0, 0, 0, 0};

  auto coop_x = [&](const float* src) {
    for (int e = threadIdx.x * 4; e < TT * DD; e += 1024) {
      const float4 f = *(const float4*)(src + e);
      union { unsigned short u[4]; uint2 q; } t;
      t.u[0] = f2bf(f.x); t.u[1] = f2bf(f.y); t.u[2] = f2bf(f.z); t.u[3] = f2bf(f.w);
      *(uint2*)(sX + e) = t.q;             // ds_store_b64
    }
  };

  v8f c[3];

  // ---- Q causal conv (shifts -2,-1,0) -> sQ row-major ----
  coop_x(q_in + base);
  __syncthreads();
  proj<3, -2>(wsQ, qb, sX, wave, lane, c);
#pragma unroll
  for (int mt = 0; mt < 3; ++mt) store_c_bf(sQ, DD, mt * 16, wave * 16, c[mt]);
  __syncthreads();

  // ---- K same conv (shifts -1,0,+1) -> per-head [s][32] ----
  coop_x(k_in + base);
  __syncthreads();
  proj<3, -1>(wsK, kb, sX, wave, lane, c);
#pragma unroll
  for (int mt = 0; mt < 3; ++mt) store_c_bf(sKT + wave * 1536, 32, mt * 16, 0, c[mt]);
  __syncthreads();

  // ---- V projection -> per-head transposed [n][64] ----
  coop_x(v_in + base);
  __syncthreads();
  proj<1, 0>(wsV, vb, sX, wave, lane, c);
#pragma unroll
  for (int mt = 0; mt < 3; ++mt) store_c_vT(sVT + (wave << 10), mt * 16, c[mt]);
  __syncthreads();                         // sQ/sKT/sVT ready; sX dead

  // ---- attention: wave w handles head h = w ----
  {
    const unsigned short* skt = sKT + wave * 1536;
    const unsigned short* svt = sVT + (wave << 10);
    unsigned short* sPw = sP + wave * (TT * 64);

    v8f s[3][3];
#pragma unroll
    for (int mt = 0; mt < 3; ++mt)
#pragma unroll
      for (int ct = 0; ct < 3; ++ct)
#pragma unroll
        for (int i = 0; i < 8; ++i) s[mt][ct][i] = 0.0f;

#pragma unroll
    for (int mt = 0; mt < 3; ++mt) {
      v16bf af = load_a_k16(sQ, DD, mt * 16, wave * 16);
#pragma unroll
      for (int ct = 0; ct < 3; ++ct) {
        v16bf bf = load_b_kT(skt, ct * 16);
        s[mt][ct] = __builtin_amdgcn_wmma_f32_16x16x32_bf16(false, af, false, bf, (short)0, s[mt][ct], false, false);
      }
    }

    // masked softmax on C layout: lane = score column (16-lane half), vgpr i = row
    const int mh = (lane & 16) ? 8 : 0;
    const int sn = lane & 15;
#pragma unroll
    for (int mt = 0; mt < 3; ++mt) {
#pragma unroll
      for (int i = 0; i < 8; ++i) {
        const int t = mt * 16 + mh + i;
        float v0 = s[mt][0][i] * 0.25f; if ( 0 + sn > t) v0 = -1.0e9f;
        float v1 = s[mt][1][i] * 0.25f; if (16 + sn > t) v1 = -1.0e9f;
        float v2 = s[mt][2][i] * 0.25f; if (32 + sn > t) v2 = -1.0e9f;
        float mx = fmaxf(fmaxf(v0, v1), v2);
#pragma unroll
        for (int d = 1; d <= 8; d <<= 1) mx = fmaxf(mx, __shfl_xor(mx, d, 32));
        float e0 = __expf(v0 - mx), e1 = __expf(v1 - mx), e2 = __expf(v2 - mx);
        float sm = e0 + e1 + e2;
#pragma unroll
        for (int d = 1; d <= 8; d <<= 1) sm += __shfl_xor(sm, d, 32);
        const float inv = 1.0f / sm;
        sPw[(t << 6) +  0 + sn] = f2bf(e0 * inv);
        sPw[(t << 6) + 16 + sn] = f2bf(e1 * inv);
        sPw[(t << 6) + 32 + sn] = f2bf(e2 * inv);
        sPw[(t << 6) + 48 + sn] = 0;       // K pad 48..63
      }
    }

    // x_h = P @ V_h  (K = 48 padded to 64)
    v8f x[3];
#pragma unroll
    for (int mt = 0; mt < 3; ++mt)
#pragma unroll
      for (int i = 0; i < 8; ++i) x[mt][i] = 0.0f;
#pragma unroll
    for (int kc = 0; kc < 2; ++kc) {
      v16bf bf = load_b_vT(svt, kc * 32);
#pragma unroll
      for (int mt = 0; mt < 3; ++mt) {
        v16bf af = load_a_p64(sPw, mt * 16, kc * 32);
        x[mt] = __builtin_amdgcn_wmma_f32_16x16x32_bf16(false, af, false, bf, (short)0, x[mt], false, false);
      }
    }
    // stash x into this head's columns of sQ (own q fully consumed)
#pragma unroll
    for (int mt = 0; mt < 3; ++mt) store_c_bf(sQ, DD, mt * 16, wave * 16, x[mt]);
  }
  __syncthreads();                         // all x visible

  // ---- output projection: out = x @ o_w + o_b ----
  {
    const float bv = ob[wave * 16 + (lane & 15)];
#pragma unroll
    for (int mt = 0; mt < 3; ++mt)
#pragma unroll
      for (int i = 0; i < 8; ++i) c[mt][i] = bv;
#pragma unroll
    for (int kc = 0; kc < 4; ++kc) {
      v16bf bf = load_b_g(wsO, kc, wave);
#pragma unroll
      for (int mt = 0; mt < 3; ++mt) {
        v16bf af = load_a_row(sQ, DD, mt * 16, kc * 32);
        c[mt] = __builtin_amdgcn_wmma_f32_16x16x32_bf16(false, af, false, bf, (short)0, c[mt], false, false);
      }
    }
    const int mh = (lane & 16) ? 8 : 0;
    const int n  = wave * 16 + (lane & 15);
#pragma unroll
    for (int mt = 0; mt < 3; ++mt)
#pragma unroll
      for (int i = 0; i < 8; ++i)
        out[base + (long)(mt * 16 + mh + i) * DD + n] = c[mt][i];
  }
}

extern "C" void kernel_launch(void* const* d_in, const int* in_sizes, int n_in,
                              void* d_out, int out_size, void* d_ws, size_t ws_size,
                              hipStream_t stream) {
  (void)in_sizes; (void)n_in; (void)out_size; (void)ws_size;
  const float* q  = (const float*)d_in[0];
  const float* k  = (const float*)d_in[1];
  const float* v  = (const float*)d_in[2];
  // d_in[3] = mask: tril(T,T) broadcast -> causal, hardcoded in-kernel
  const float* qw = (const float*)d_in[4];
  const float* qb = (const float*)d_in[5];
  const float* kw = (const float*)d_in[6];
  const float* kb = (const float*)d_in[7];
  const float* vw = (const float*)d_in[8];
  const float* vb = (const float*)d_in[9];
  const float* ow = (const float*)d_in[10];
  const float* ob = (const float*)d_in[11];
  float* out = (float*)d_out;
  unsigned short* wsw = (unsigned short*)d_ws;   // 8 * 16384 bf16 = 256KB

  swizzle_weights_kernel<<<dim3((8 * DD * DD + 255) / 256), 256, 0, stream>>>(qw, kw, vw, ow, wsw);
  fused_attn_kernel<<<dim3(8 * 512), 256, 0, stream>>>(q, k, v, wsw, qb, kb, vb, ob, out);
}
